// MultiHeadAttention_20203526160437
// MI455X (gfx1250) — compile-verified, matrix operand order checked by reading, not run on hardware
//
#include <hip/hip_runtime.h>

// ---------------------------------------------------------------------------
// MHA on gfx1250: f32 -> f16 staging, all matmuls via v_wmma_f32_16x16x32_f16,
// flash-style attention with online softmax.
// ---------------------------------------------------------------------------

typedef __attribute__((ext_vector_type(16))) _Float16 v16h;
typedef __attribute__((ext_vector_type(8)))  float    v8f;
typedef __attribute__((ext_vector_type(4)))  float    f32x4;

#define D_MODEL 1024
#define N_HEADS 16
#define DKH     64
#define BATCH   4
#define SEQ     2048

// Load one 16-lane-striped f16 WMMA operand fragment (A: 16xK rows, or B for
// an NT gemm where the B source is row-major N x K).  Lane striping per ISA:
// lanes 0-15 hold k = {base..base+7, base+16..base+23}, lanes 16-31 hold
// k = {base+8..base+15, base+24..base+31}; caller folds the +8 into `p`.
__device__ __forceinline__ v16h ldfrag(const _Float16* p) {
  union { v16h v; f32x4 f[2]; } u;
  u.f[0] = *(const f32x4*)(p);
  u.f[1] = *(const f32x4*)(p + 16);
  return u.v;
}

__device__ __forceinline__ v8f wmma16(v16h a, v16h b, v8f c) {
  return __builtin_amdgcn_wmma_f32_16x16x32_f16(
      /*neg_a=*/false, a, /*neg_b=*/false, b,
      /*c_mod=*/(short)0, c, /*reuse_a=*/false, /*reuse_b=*/false);
}

// Reductions across a 16-lane group (xor masks 1,2,4,8 stay inside halves).
__device__ __forceinline__ float rmax16(float v) {
  v = fmaxf(v, __shfl_xor(v, 1, 32));
  v = fmaxf(v, __shfl_xor(v, 2, 32));
  v = fmaxf(v, __shfl_xor(v, 4, 32));
  v = fmaxf(v, __shfl_xor(v, 8, 32));
  return v;
}
__device__ __forceinline__ float rsum16(float v) {
  v += __shfl_xor(v, 1, 32);
  v += __shfl_xor(v, 2, 32);
  v += __shfl_xor(v, 4, 32);
  v += __shfl_xor(v, 8, 32);
  return v;
}

// ---------------------------------------------------------------------------
__global__ void cvt_f32_f16(const float* __restrict__ s, _Float16* __restrict__ d, int n) {
  int i = blockIdx.x * blockDim.x + threadIdx.x;
  int stride = gridDim.x * blockDim.x;
  for (; i < n; i += stride) d[i] = (_Float16)s[i];
}

// ---------------------------------------------------------------------------
// C[M,N] = A[M,K] * W[N,K]^T + bias   (NT gemm, both operands row-major in K)
// Block tile 128x128, 8 waves (4 along M x 2 along N), 32x64 per wave.
__global__ __launch_bounds__(256) void gemm_nt(
    const _Float16* __restrict__ A, const _Float16* __restrict__ W,
    const float* __restrict__ bias, float* __restrict__ Cf32,
    _Float16* __restrict__ Cf16, int M, int N, int K) {
  const int lane  = threadIdx.x & 31;
  const int w     = threadIdx.x >> 5;
  const int wm    = w & 3;
  const int wn    = w >> 2;
  const int row0  = blockIdx.y * 128 + wm * 32;
  const int col0  = blockIdx.x * 128 + wn * 64;
  const int r     = lane & 15;
  const int khalf = (lane >> 4) << 3;

  v8f acc[2][4] = {};

  const _Float16* arow = A + (size_t)(row0 + r) * K + khalf;
  const _Float16* wrow = W + (size_t)(col0 + r) * K + khalf;

  for (int kk = 0; kk < K; kk += 32) {
    __builtin_prefetch(arow + kk + 64, 0, 0);      // global_prefetch_b8
    __builtin_prefetch(wrow + kk + 64, 0, 0);
    v16h a[2], b[4];
#pragma unroll
    for (int mt = 0; mt < 2; ++mt)
      a[mt] = ldfrag(arow + (size_t)(mt * 16) * K + kk);
#pragma unroll
    for (int nt = 0; nt < 4; ++nt)
      b[nt] = ldfrag(wrow + (size_t)(nt * 16) * K + kk);
#pragma unroll
    for (int mt = 0; mt < 2; ++mt)
#pragma unroll
      for (int nt = 0; nt < 4; ++nt)
        acc[mt][nt] = wmma16(a[mt], b[nt], acc[mt][nt]);
  }

  // C layout: vgpr j -> row j (lanes 0-15) / row j+8 (lanes 16-31), col = lane&15
  const int rbase = (lane >> 4) << 3;
  const int col   = lane & 15;
#pragma unroll
  for (int mt = 0; mt < 2; ++mt)
#pragma unroll
    for (int nt = 0; nt < 4; ++nt)
#pragma unroll
      for (int j = 0; j < 8; ++j) {
        int rr = row0 + mt * 16 + rbase + j;
        int cc = col0 + nt * 16 + col;
        float v = acc[mt][nt][j] + bias[cc];
        if (Cf32) Cf32[(size_t)rr * N + cc] = v;
        else      Cf16[(size_t)rr * N + cc] = (_Float16)v;
      }
}

// ---------------------------------------------------------------------------
// Flash attention: one workgroup per (b, h, 128-row block). 8 waves x 16 rows.
// Per 128-col block: S = Q*K^T (WMMA), online softmax, P via LDS re-striping,
// O += P*V (WMMA, V transposed into LDS).
__global__ __launch_bounds__(256) void attn_kernel(
    const _Float16* __restrict__ qp, const _Float16* __restrict__ kp,
    const _Float16* __restrict__ vp, _Float16* __restrict__ ao) {
  const int b = blockIdx.z, h = blockIdx.y, rb = blockIdx.x;
  const int w     = threadIdx.x >> 5;
  const int lane  = threadIdx.x & 31;
  const int r     = lane & 15;
  const int khalf = (lane >> 4) << 3;
  const int rbase = (lane >> 4) << 3;
  const int col   = lane & 15;

  __shared__ __align__(16) _Float16 Vt[DKH][136];        // V^T tile: [dk][col]
  __shared__ __align__(16) _Float16 Psh[8][16][136];     // per-wave P tile

  const size_t bOff = (size_t)b * SEQ * D_MODEL;
  const int    hcol = h * DKH;
  const int    row0 = rb * 128 + w * 16;

  // Preload Q fragments (16 rows x 64 dk = two K=32 steps)
  v16h qf[2];
#pragma unroll
  for (int ks = 0; ks < 2; ++ks)
    qf[ks] = ldfrag(qp + bOff + (size_t)(row0 + r) * D_MODEL + hcol + ks * 32 + khalf);

  v8f accO[4] = {};
  float m[8], l[8];
#pragma unroll
  for (int j = 0; j < 8; ++j) { m[j] = -1e30f; l[j] = 0.0f; }

  for (int cb = 0; cb < SEQ / 128; ++cb) {
    __syncthreads();  // prev iteration's Vt readers done

    // Cooperative transpose of V block into LDS: Vt[dk][c] = V[cb*128+c][dk]
    {
      const int c  = threadIdx.x & 127;
      const int hs = threadIdx.x >> 7;  // 0/1 -> dk halves of 32
      const _Float16* src = vp + bOff + (size_t)(cb * 128 + c) * D_MODEL + hcol + hs * 32;
#pragma unroll
      for (int i4 = 0; i4 < 4; ++i4) {
        union { f32x4 f; _Float16 hh[8]; } u;
        u.f = *(const f32x4*)(src + i4 * 8);
#pragma unroll
        for (int e = 0; e < 8; ++e) Vt[hs * 32 + i4 * 8 + e][c] = u.hh[e];
      }
    }

    // Scores: S[16 x 128] = Q_tile * K_block^T
    v8f s[8];
    const v8f z = {};
#pragma unroll
    for (int t = 0; t < 8; ++t) {
      s[t] = z;
#pragma unroll
      for (int ks = 0; ks < 2; ++ks) {
        v16h bf = ldfrag(kp + bOff + (size_t)(cb * 128 + t * 16 + r) * D_MODEL +
                         hcol + ks * 32 + khalf);
        s[t] = wmma16(qf[ks], bf, s[t]);
      }
    }

    // Online softmax update
    const float sc = 0.125f;  // 1/sqrt(64)
    float fac[8];
#pragma unroll
    for (int t = 0; t < 8; ++t)
#pragma unroll
      for (int j = 0; j < 8; ++j) s[t][j] *= sc;
#pragma unroll
    for (int j = 0; j < 8; ++j) {
      float mx = -1e30f;
#pragma unroll
      for (int t = 0; t < 8; ++t) mx = fmaxf(mx, s[t][j]);
      mx = rmax16(mx);
      float mn = fmaxf(m[j], mx);
      fac[j] = __expf(m[j] - mn);
      m[j] = mn;
    }
#pragma unroll
    for (int j = 0; j < 8; ++j) {
      float rs = 0.0f;
#pragma unroll
      for (int t = 0; t < 8; ++t) {
        float p = __expf(s[t][j] - m[j]);
        s[t][j] = p;
        rs += p;
      }
      rs = rsum16(rs);
      l[j] = l[j] * fac[j] + rs;
    }
#pragma unroll
    for (int ot = 0; ot < 4; ++ot)
#pragma unroll
      for (int j = 0; j < 8; ++j) accO[ot][j] *= fac[j];

    // Re-stripe P (C layout -> A layout) through wave-private LDS
#pragma unroll
    for (int t = 0; t < 8; ++t)
#pragma unroll
      for (int j = 0; j < 8; ++j)
        Psh[w][rbase + j][t * 16 + col] = (_Float16)s[t][j];

    __syncthreads();  // Vt ready for all waves

    // O += P[16x128] * V[128x64]
#pragma unroll
    for (int ks = 0; ks < 4; ++ks) {
      v16h pf = ldfrag(&Psh[w][r][ks * 32 + khalf]);
#pragma unroll
      for (int ot = 0; ot < 4; ++ot) {
        v16h vf = ldfrag(&Vt[ot * 16 + r][ks * 32 + khalf]);
        accO[ot] = wmma16(pf, vf, accO[ot]);
      }
    }
  }

  // Epilogue: normalize and write combined [B,S,H*Dk] f16
#pragma unroll
  for (int ot = 0; ot < 4; ++ot)
#pragma unroll
    for (int j = 0; j < 8; ++j) {
      float o = accO[ot][j] / l[j];
      ao[bOff + (size_t)(row0 + rbase + j) * D_MODEL + hcol + ot * 16 + col] = (_Float16)o;
    }
}

// ---------------------------------------------------------------------------
extern "C" void kernel_launch(void* const* d_in, const int* in_sizes, int n_in,
                              void* d_out, int out_size, void* d_ws, size_t ws_size,
                              hipStream_t stream) {
  (void)in_sizes; (void)n_in; (void)out_size; (void)ws_size;
  const size_t BS  = (size_t)BATCH * SEQ;       // 8192
  const size_t BSD = BS * D_MODEL;              // 8,388,608
  const size_t DD  = (size_t)D_MODEL * D_MODEL; // 1,048,576

  const float* Q  = (const float*)d_in[0];
  const float* K  = (const float*)d_in[1];
  const float* V  = (const float*)d_in[2];
  const float* Wq = (const float*)d_in[3];
  const float* bq = (const float*)d_in[4];
  const float* Wk = (const float*)d_in[5];
  const float* bk = (const float*)d_in[6];
  const float* Wv = (const float*)d_in[7];
  const float* bv = (const float*)d_in[8];
  const float* Wo = (const float*)d_in[9];
  const float* bo = (const float*)d_in[10];

  _Float16* ws  = (_Float16*)d_ws;
  _Float16* hQ  = ws;
  _Float16* hK  = hQ  + BSD;
  _Float16* hV  = hK  + BSD;
  _Float16* hWq = hV  + BSD;
  _Float16* hWk = hWq + DD;
  _Float16* hWv = hWk + DD;
  _Float16* hWo = hWv + DD;
  _Float16* qp  = hWo + DD;
  _Float16* kp  = qp  + BSD;
  _Float16* vpj = kp  + BSD;
  _Float16* ao  = vpj + BSD;

  // f32 -> f16 staging
  cvt_f32_f16<<<2048, 256, 0, stream>>>(Q,  hQ,  (int)BSD);
  cvt_f32_f16<<<2048, 256, 0, stream>>>(K,  hK,  (int)BSD);
  cvt_f32_f16<<<2048, 256, 0, stream>>>(V,  hV,  (int)BSD);
  cvt_f32_f16<<<1024, 256, 0, stream>>>(Wq, hWq, (int)DD);
  cvt_f32_f16<<<1024, 256, 0, stream>>>(Wk, hWk, (int)DD);
  cvt_f32_f16<<<1024, 256, 0, stream>>>(Wv, hWv, (int)DD);
  cvt_f32_f16<<<1024, 256, 0, stream>>>(Wo, hWo, (int)DD);

  // Projections (NT gemm, y = x @ W^T + b), f16 output
  dim3 gg(D_MODEL / 128, (unsigned)(BS / 128));
  gemm_nt<<<gg, 256, 0, stream>>>(hQ, hWq, bq, nullptr, qp, (int)BS, D_MODEL, D_MODEL);
  gemm_nt<<<gg, 256, 0, stream>>>(hK, hWk, bk, nullptr, kp, (int)BS, D_MODEL, D_MODEL);
  gemm_nt<<<gg, 256, 0, stream>>>(hV, hWv, bv, nullptr, vpj, (int)BS, D_MODEL, D_MODEL);

  // Attention
  attn_kernel<<<dim3(SEQ / 128, N_HEADS, BATCH), 256, 0, stream>>>(qp, kp, vpj, ao);

  // Output projection, f32 output
  gemm_nt<<<gg, 256, 0, stream>>>(ao, hWo, bo, (float*)d_out, nullptr, (int)BS, D_MODEL, D_MODEL);
}
